// GraphAttentionLayer_25709674234623
// MI455X (gfx1250) — compile-verified, matrix operand order checked by reading, not run on hardware
//
#include <hip/hip_runtime.h>
#include <math.h>

#define ALPHA 0.2f
#define NEG_BIG -9.0e15f
#define B_  16
#define N_  1024
#define F_  256

typedef __bf16          bf16x16 __attribute__((ext_vector_type(16)));
typedef float           f32x8   __attribute__((ext_vector_type(8)));
typedef unsigned short  u16x8   __attribute__((ext_vector_type(8)));
typedef unsigned short  u16x16  __attribute__((ext_vector_type(16)));

// ---------- helpers ----------
__device__ __forceinline__ unsigned short f2bf(float f) {
    unsigned int u = __builtin_bit_cast(unsigned int, f);
    u += 0x7FFFu + ((u >> 16) & 1u);          // round-to-nearest-even
    return (unsigned short)(u >> 16);
}
__device__ __forceinline__ float bf2f(unsigned short s) {
    unsigned int u = ((unsigned int)s) << 16;
    return __builtin_bit_cast(float, u);
}

// Per-lane A/B fragment for v_wmma_f32_16x16x32_bf16.
// rowPtr points at the lane's (row|col = lane&15) line; fragment element i
// holds k = (i>>3)*16 + laneHalf*8 + (i&7)   (ISA 7.12.2 16-bit A layout;
// B mirrors it with N in place of M).  Two 16-byte loads per fragment.
__device__ __forceinline__ bf16x16 make_frag(const unsigned short* rowPtr,
                                             int kBase, int laneHalf) {
    const u16x8* p0 = (const u16x8*)(rowPtr + kBase + laneHalf * 8);
    const u16x8* p1 = (const u16x8*)(rowPtr + kBase + 16 + laneHalf * 8);
    u16x8 lo = *p0;
    u16x8 hi = *p1;
    u16x16 full;
#pragma unroll
    for (int i = 0; i < 8; ++i) { full[i] = lo[i]; full[8 + i] = hi[i]; }
    return __builtin_bit_cast(bf16x16, full);
}

// ---------- stage 0: fp32 -> bf16 conversions ----------
__global__ void k_cvt_h(const float* __restrict__ h, unsigned short* __restrict__ hb) {
    size_t i = (size_t)blockIdx.x * blockDim.x + threadIdx.x;   // B*N*F threads
    hb[i] = f2bf(h[i]);
}
// Wt[o][k] = W[k][o]  (bf16): B-fragments of GEMM1 become k-contiguous per lane
__global__ void k_cvt_wt(const float* __restrict__ W, unsigned short* __restrict__ wt) {
    int i = blockIdx.x * blockDim.x + threadIdx.x;              // F*F threads
    int o = i >> 8, k = i & 255;
    wt[i] = f2bf(W[k * F_ + o]);
}

// ---------- stage 1: WhT[b][o][n] = (h @ W)^T, bf16, WMMA ----------
// 2x2 register blocking: each wave computes a 32(row) x 32(o) block.
// 4 accumulators + 4 fragments/step: no spill pressure, 8 b128 : 4 wmma.
__global__ void __launch_bounds__(256)
k_gemm_wh(const unsigned short* __restrict__ hb,
          const unsigned short* __restrict__ wt,
          unsigned short* __restrict__ whT) {
    int wave  = (blockIdx.x * blockDim.x + threadIdx.x) >> 5;   // 4096 waves
    int lane  = threadIdx.x & 31;
    int l16   = lane & 15;
    int lHalf = (lane >> 4) & 1;
    int oB    = wave & 7;            // 8 o-blocks of 32
    int tM2   = wave >> 3;           // 512 row-blocks of 32 over B*N

    const unsigned short* aRow0 = hb + (size_t)(tM2 * 32 + l16) * F_;
    const unsigned short* aRow1 = aRow0 + (size_t)16 * F_;
    const unsigned short* bRow0 = wt + (size_t)(oB * 32 + l16) * F_;
    const unsigned short* bRow1 = bRow0 + (size_t)16 * F_;

    f32x8 c00 = {}, c01 = {}, c10 = {}, c11 = {};
#pragma unroll
    for (int k = 0; k < F_; k += 32) {
        bf16x16 a0 = make_frag(aRow0, k, lHalf);
        bf16x16 a1 = make_frag(aRow1, k, lHalf);
        bf16x16 b0 = make_frag(bRow0, k, lHalf);
        bf16x16 b1 = make_frag(bRow1, k, lHalf);
        c00 = __builtin_amdgcn_wmma_f32_16x16x32_bf16(false, a0, false, b0, (short)0, c00, false, false);
        c01 = __builtin_amdgcn_wmma_f32_16x16x32_bf16(false, a0, false, b1, (short)0, c01, false, false);
        c10 = __builtin_amdgcn_wmma_f32_16x16x32_bf16(false, a1, false, b0, (short)0, c10, false, false);
        c11 = __builtin_amdgcn_wmma_f32_16x16x32_bf16(false, a1, false, b1, (short)0, c11, false, false);
    }
    // D: VGPR r -> (M = r + 8*laneHalf, N = lane&15).  Store transposed:
    // WhT[b][o][n]; consecutive r -> consecutive n -> one 16-byte store each.
    f32x8 acc[2][2] = {{c00, c01}, {c10, c11}};
#pragma unroll
    for (int m = 0; m < 2; ++m) {
        int gRow = tM2 * 32 + m * 16;
        int bb   = gRow >> 10;                       // 32-row blocks never cross batch
        int nIn  = (gRow & (N_ - 1)) + lHalf * 8;
#pragma unroll
        for (int t = 0; t < 2; ++t) {
            int o = oB * 32 + t * 16 + l16;
            u16x8 outv;
#pragma unroll
            for (int r = 0; r < 8; ++r) outv[r] = f2bf(acc[m][t][r]);
            *(u16x8*)(whT + ((size_t)bb * F_ + o) * N_ + nIn) = outv;
        }
    }
}

// ---------- stage 2: f1[b,n] = Wh[b,n,:]@a1 ; f2 likewise ----------
__global__ void k_f12(const unsigned short* __restrict__ whT,
                      const float* __restrict__ a,
                      float* __restrict__ f1, float* __restrict__ f2) {
    int idx = blockIdx.x * blockDim.x + threadIdx.x;            // B*N threads
    int bb  = idx >> 10, n = idx & (N_ - 1);
    const unsigned short* base = whT + (size_t)bb * F_ * N_ + n;
    float s1 = 0.f, s2 = 0.f;
#pragma unroll 4
    for (int o = 0; o < F_; ++o) {
        float w = bf2f(base[(size_t)o * N_]);                   // coalesced in n
        s1 += w * a[o];
        s2 += w * a[o + F_];
    }
    f1[idx] = s1;
    f2[idx] = s2;
}

// ---------- stage 3: masked softmax row -> bf16 attention ----------
__global__ void k_softmax(const float* __restrict__ f1, const float* __restrict__ f2,
                          const int* __restrict__ adj, unsigned short* __restrict__ att) {
    __shared__ float red[256];
    int row = blockIdx.x;                    // b*N + i
    int bb  = row >> 10;
    int tid = threadIdx.x;
    float f1v = f1[row];

    float sv[4];
#pragma unroll
    for (int k = 0; k < 4; ++k) {
        int   j = tid + k * 256;
        // stream-prefetch the adj row ahead of use (global_prefetch_b8)
        __builtin_prefetch(adj + (size_t)row * N_ + j, 0, 1);
        float e = f1v + f2[bb * N_ + j];
        e = (e > 0.f) ? e : ALPHA * e;       // leaky_relu before mask (ref order)
        int ad = adj[(size_t)row * N_ + j];
        sv[k] = (ad > 0) ? e : NEG_BIG;
    }
    float m = fmaxf(fmaxf(sv[0], sv[1]), fmaxf(sv[2], sv[3]));
    red[tid] = m; __syncthreads();
    for (int s = 128; s > 0; s >>= 1) {
        if (tid < s) red[tid] = fmaxf(red[tid], red[tid + s]);
        __syncthreads();
    }
    m = red[0]; __syncthreads();

    float pv[4], sum = 0.f;
#pragma unroll
    for (int k = 0; k < 4; ++k) { pv[k] = __expf(sv[k] - m); sum += pv[k]; }
    red[tid] = sum; __syncthreads();
    for (int s = 128; s > 0; s >>= 1) {
        if (tid < s) red[tid] += red[tid + s];
        __syncthreads();
    }
    float inv = 1.0f / red[0];
#pragma unroll
    for (int k = 0; k < 4; ++k) {
        int j = tid + k * 256;
        att[(size_t)row * N_ + j] = f2bf(pv[k] * inv);
    }
}

// ---------- stage 4: out = elu(att @ Wh), WMMA, 2x2 register blocking ----------
__global__ void __launch_bounds__(256)
k_gemm_out(const unsigned short* __restrict__ att,
           const unsigned short* __restrict__ whT,
           float* __restrict__ out) {
    int wave  = (blockIdx.x * blockDim.x + threadIdx.x) >> 5;   // 4096 waves
    int lane  = threadIdx.x & 31;
    int l16   = lane & 15;
    int lHalf = (lane >> 4) & 1;
    int bb    = wave >> 8;           // 16 batches
    int iB    = (wave >> 3) & 31;    // 32 i-blocks of 32
    int oB    = wave & 7;            // 8 o-blocks of 32

    const unsigned short* aRow0 = att + ((size_t)bb * N_ + iB * 32 + l16) * N_;
    const unsigned short* aRow1 = aRow0 + (size_t)16 * N_;
    const unsigned short* bRow0 = whT + ((size_t)bb * F_ + oB * 32 + l16) * N_;
    const unsigned short* bRow1 = bRow0 + (size_t)16 * N_;

    f32x8 c00 = {}, c01 = {}, c10 = {}, c11 = {};
    for (int k = 0; k < N_; k += 32) {       // 32 k-steps, 8 b128 : 4 wmma
        bf16x16 a0 = make_frag(aRow0, k, lHalf);
        bf16x16 a1 = make_frag(aRow1, k, lHalf);
        bf16x16 b0 = make_frag(bRow0, k, lHalf);
        bf16x16 b1 = make_frag(bRow1, k, lHalf);
        c00 = __builtin_amdgcn_wmma_f32_16x16x32_bf16(false, a0, false, b0, (short)0, c00, false, false);
        c01 = __builtin_amdgcn_wmma_f32_16x16x32_bf16(false, a0, false, b1, (short)0, c01, false, false);
        c10 = __builtin_amdgcn_wmma_f32_16x16x32_bf16(false, a1, false, b0, (short)0, c10, false, false);
        c11 = __builtin_amdgcn_wmma_f32_16x16x32_bf16(false, a1, false, b1, (short)0, c11, false, false);
    }
    f32x8 acc[2][2] = {{c00, c01}, {c10, c11}};
#pragma unroll
    for (int m = 0; m < 2; ++m) {
        int iBase = iB * 32 + m * 16 + lHalf * 8;
#pragma unroll
        for (int t = 0; t < 2; ++t) {
            int o = oB * 32 + t * 16 + l16;
#pragma unroll
            for (int r = 0; r < 8; ++r) {
                float v = acc[m][t][r];
                v = (v > 0.f) ? v : (__expf(v) - 1.0f);         // elu, alpha=1
                out[((size_t)bb * N_ + iBase + r) * F_ + o] = v;
            }
        }
    }
}

// ---------- host ----------
extern "C" void kernel_launch(void* const* d_in, const int* in_sizes, int n_in,
                              void* d_out, int out_size, void* d_ws, size_t ws_size,
                              hipStream_t stream) {
    const float* h   = (const float*)d_in[0];   // B*N*F fp32
    const int*   adj = (const int*)d_in[1];     // B*N*N int32
    const float* W   = (const float*)d_in[2];   // F*F   fp32
    const float* a   = (const float*)d_in[3];   // 2F    fp32
    float* out = (float*)d_out;

    char* ws = (char*)d_ws;
    unsigned short* hb  = (unsigned short*)(ws);                         //  8 MiB
    unsigned short* wt  = (unsigned short*)(ws + 8388608);               // 128 KiB
    unsigned short* whT = (unsigned short*)(ws + 8519680);               //  8 MiB
    float*          f1  = (float*)(ws + 16908288);                       //  64 KiB
    float*          f2  = (float*)(ws + 16973824);                       //  64 KiB
    unsigned short* att = (unsigned short*)(ws + 17039360);              // 32 MiB

    const size_t elems = (size_t)B_ * N_ * F_;                           // 4M

    k_cvt_h   <<<elems / 256, 256, 0, stream>>>(h, hb);
    k_cvt_wt  <<<(F_ * F_) / 256, 256, 0, stream>>>(W, wt);
    k_gemm_wh <<<512, 256, 0, stream>>>(hb, wt, whT);     // 4096 waves, 32x32 blocks
    k_f12     <<<(B_ * N_) / 256, 256, 0, stream>>>(whT, a, f1, f2);
    k_softmax <<<B_ * N_, 256, 0, stream>>>(f1, f2, adj, att);
    k_gemm_out<<<512, 256, 0, stream>>>(att, whT, out);   // 4096 waves, 32x32 blocks
}
